// VoxelSetAbstraction_59407987638871
// MI455X (gfx1250) — matrix operand order repaired and checked by reading, not compiled, not run on hardware
//
#include <hip/hip_runtime.h>
#include <hip/hip_bf16.h>

typedef __attribute__((ext_vector_type(16))) _Float16 v16h;
typedef __attribute__((ext_vector_type(8)))  _Float16 v8h;
typedef __attribute__((ext_vector_type(8)))  float    v8f;

#define N_PTS   16384
#define N_KP    4096
#define FPS_THREADS 1024
#define FPS_PPT 16          // 16384 / 1024
#define OUT_STRIDE 131      // 3 + 128

// ---------------------------------------------------------------------------
// 1) Furthest point sampling: one block per batch. Points live in registers
//    (strided ownership p = t + j*1024). 3-stage argmax reduction per step.
// ---------------------------------------------------------------------------
__global__ __launch_bounds__(FPS_THREADS)
void fps_kernel(const float* __restrict__ clouds, int N, int K,
                int* __restrict__ kp_idx, float* __restrict__ kp_xyz) {
  const int b = blockIdx.x;
  const int t = threadIdx.x;
  const int lane = t & 31, wid = t >> 5;
  const float* cx = clouds + (size_t)b * 4 * N;
  const float* cy = cx + N;
  const float* cz = cx + 2 * N;

  float px[FPS_PPT], py[FPS_PPT], pz[FPS_PPT], dmin[FPS_PPT];
#pragma unroll
  for (int j = 0; j < FPS_PPT; ++j) {
    int p = t + j * FPS_THREADS;
    px[j] = cx[p]; py[j] = cy[p]; pz[j] = cz[p];
    dmin[j] = 1e10f;
  }

  __shared__ float s_bx, s_by, s_bz;
  __shared__ float s_wv[32];
  __shared__ int   s_wi[32];
  __shared__ int   s_next;

  int last = 0;
  for (int i = 0; i < K; ++i) {
    if (t == (last & (FPS_THREADS - 1))) {
      int j = last >> 10;
      s_bx = px[j]; s_by = py[j]; s_bz = pz[j];
      kp_idx[(size_t)b * K + i] = last;
      float* o = kp_xyz + ((size_t)b * K + i) * 3;
      o[0] = px[j]; o[1] = py[j]; o[2] = pz[j];
    }
    __syncthreads();
    const float bx = s_bx, by = s_by, bz = s_bz;

    float bestv = -1.0f; int besti = 0x7fffffff;
#pragma unroll
    for (int j = 0; j < FPS_PPT; ++j) {
      float dx = px[j] - bx, dy = py[j] - by, dz = pz[j] - bz;
      float d = dx * dx + dy * dy + dz * dz;
      float m = fminf(dmin[j], d);
      dmin[j] = m;
      int p = t + j * FPS_THREADS;
      if (m > bestv || (m == bestv && p < besti)) { bestv = m; besti = p; }
    }
    // wave32 reduce
#pragma unroll
    for (int off = 16; off > 0; off >>= 1) {
      float ov = __shfl_down(bestv, off, 32);
      int   oi = __shfl_down(besti, off, 32);
      if (ov > bestv || (ov == bestv && oi < besti)) { bestv = ov; besti = oi; }
    }
    if (lane == 0) { s_wv[wid] = bestv; s_wi[wid] = besti; }
    __syncthreads();
    if (wid == 0) {
      bestv = s_wv[lane]; besti = s_wi[lane];
#pragma unroll
      for (int off = 16; off > 0; off >>= 1) {
        float ov = __shfl_down(bestv, off, 32);
        int   oi = __shfl_down(besti, off, 32);
        if (ov > bestv || (ov == bestv && oi < besti)) { bestv = ov; besti = oi; }
      }
      if (lane == 0) s_next = besti;
    }
    __syncthreads();
    last = s_next;
  }
}

// ---------------------------------------------------------------------------
// 2) Top-32 nearest neighbors per keypoint, sorted ascending by d2.
//    One thread per keypoint; register-resident insertion sort; point tiles
//    staged through LDS. Top-16 prefix serves the r=0.8 scale.
// ---------------------------------------------------------------------------
__global__ __launch_bounds__(256)
void knn_kernel(const float* __restrict__ clouds, const float* __restrict__ kp_xyz,
                int N, int K, int* __restrict__ neigh_idx, float* __restrict__ neigh_d2) {
  const int gid = blockIdx.x * 256 + threadIdx.x;     // keypoint id in [0, B*K)
  const int b = gid / K;
  const float* cx = clouds + (size_t)b * 4 * N;
  const float* cy = cx + N;
  const float* cz = cx + 2 * N;

  const float kx = kp_xyz[(size_t)gid * 3 + 0];
  const float ky = kp_xyz[(size_t)gid * 3 + 1];
  const float kz = kp_xyz[(size_t)gid * 3 + 2];

  float bd[32]; int bi[32];
#pragma unroll
  for (int j = 0; j < 32; ++j) { bd[j] = 3.4e38f; bi[j] = 0; }

  __shared__ float tx[256], ty[256], tz[256];
  for (int t0 = 0; t0 < N; t0 += 256) {
    __syncthreads();
    int p = t0 + threadIdx.x;
    tx[threadIdx.x] = cx[p]; ty[threadIdx.x] = cy[p]; tz[threadIdx.x] = cz[p];
    __syncthreads();
#pragma unroll 4
    for (int jj = 0; jj < 256; ++jj) {
      float dx = tx[jj] - kx, dy = ty[jj] - ky, dz = tz[jj] - kz;
      float d = dx * dx + dy * dy + dz * dz;
      if (d < bd[31]) {
        int pidx = t0 + jj;
#pragma unroll
        for (int j = 31; j > 0; --j) {
          if (d < bd[j]) {
            if (d >= bd[j - 1]) { bd[j] = d; bi[j] = pidx; }
            else               { bd[j] = bd[j - 1]; bi[j] = bi[j - 1]; }
          }
        }
        if (d < bd[0]) { bd[0] = d; bi[0] = pidx; }
      }
    }
  }
#pragma unroll
  for (int j = 0; j < 32; ++j) {
    neigh_idx[(size_t)gid * 32 + j] = bi[j];
    neigh_d2 [(size_t)gid * 32 + j] = bd[j];
  }
}

// ---------------------------------------------------------------------------
// 3) Pack a row-major f32 weight [Kdim, Ncols] into per-lane WMMA B-fragment
//    layout (f16): frag[((ks*nct + ct)*32 + lane)*16 + e] = W[k][n],
//    k = ks*32 + (lane>>4)*16 + e, n = ct*16 + (lane&15).
// ---------------------------------------------------------------------------
__global__ void pack_w_kernel(const float* __restrict__ src, _Float16* __restrict__ dst,
                              int Kdim, int Ncols) {
  int tid = blockIdx.x * blockDim.x + threadIdx.x;
  if (tid >= Kdim * Ncols) return;
  int e    = tid & 15;
  int lane = (tid >> 4) & 31;
  int rest = tid >> 9;
  int nct  = Ncols >> 4;
  int ct   = rest % nct;
  int ks   = rest / nct;
  int k = ks * 32 + (lane >> 4) * 16 + e;
  int n = ct * 16 + (lane & 15);
  dst[tid] = (_Float16)src[(size_t)k * Ncols + n];
}

// ---------------------------------------------------------------------------
// 4) Set-abstraction per keypoint: gather -> layer1 (VALU, K=4) -> layer2
//    WMMA f16 (128 -> 256) -> masked relu -> max-pool over samples.
//    One workgroup (8 waves) per keypoint; wave owns 2 column tiles.
// ---------------------------------------------------------------------------
__global__ __launch_bounds__(256)
void sa_kernel(const float* __restrict__ clouds, const float* __restrict__ kp_xyz,
               const int* __restrict__ neigh_idx, const float* __restrict__ neigh_d2,
               const float* __restrict__ w1,          // [4,128] row-major f32
               const _Float16* __restrict__ w2_frag,  // packed B fragments
               float* __restrict__ pooled,            // [B*K, 256]
               int N, int K, int ns, float r2) {
  const int kp = blockIdx.x;               // 0 .. B*K-1
  const int b  = kp / K;
  const int t  = threadIdx.x;
  const int lane = t & 31, wid = t >> 5;
  const int hi = lane >> 4, ln = lane & 15;

  __shared__ __align__(16) _Float16 sH1[32 * 128];   // relu(g @ W1), f16
  __shared__ float sG[32 * 4];
  __shared__ float sValid[32];

  const float* cl = clouds + (size_t)b * 4 * N;
  const float kx = kp_xyz[(size_t)kp * 3 + 0];
  const float ky = kp_xyz[(size_t)kp * 3 + 1];
  const float kz = kp_xyz[(size_t)kp * 3 + 2];

  if (t < ns) {
    int   pi  = neigh_idx[(size_t)kp * 32 + t];
    float d2v = neigh_d2 [(size_t)kp * 32 + t];
    sG[t * 4 + 0] = cl[pi]         - kx;
    sG[t * 4 + 1] = cl[N + pi]     - ky;
    sG[t * 4 + 2] = cl[2 * N + pi] - kz;
    sG[t * 4 + 3] = cl[3 * N + pi];
    sValid[t] = (d2v < r2) ? 1.0f : 0.0f;
  } else if (t < 32) {
    sValid[t] = 0.0f;
  }
  __syncthreads();

  // layer1: [ns,4] @ [4,128] + relu -> f16 in LDS
  const int total = ns * 128;
  for (int o = t; o < total; o += 256) {
    int s = o >> 7, c = o & 127;
    float acc = sG[s * 4 + 0] * w1[c]
              + sG[s * 4 + 1] * w1[128 + c]
              + sG[s * 4 + 2] * w1[256 + c]
              + sG[s * 4 + 3] * w1[384 + c];
    sH1[s * 128 + c] = (_Float16)fmaxf(acc, 0.0f);
  }
  __syncthreads();

  const int NT = ns >> 4;                  // row tiles of 16 samples
  for (int cti = 0; cti < 2; ++cti) {
    const int ct = wid * 2 + cti;          // 8 waves * 2 = 16 col tiles (256 cols)
    float m = 0.0f;
    for (int rt = 0; rt < NT; ++rt) {
      v8f c = {};
#pragma unroll
      for (int ks = 0; ks < 4; ++ks) {
        // A fragment (contiguous k order, same as packed B): lane holds row m = lane&15;
        // halves 0..7 -> K = ks*32 + hi*16 + e ; halves 8..15 -> +8.
        const _Float16* ap = &sH1[((rt << 4) + ln) * 128 + ks * 32 + hi * 16];
        v8h alo = *(const v8h*)(ap);
        v8h ahi = *(const v8h*)(ap + 8);
        v16h a;
#pragma unroll
        for (int e = 0; e < 8; ++e) { a[e] = alo[e]; a[8 + e] = ahi[e]; }
        v16h bf = *(const v16h*)(w2_frag + ((size_t)((ks << 4) + ct) * 32 + lane) * 16);
        c = __builtin_amdgcn_wmma_f32_16x16x32_f16(
            false, a, false, bf, (short)0, c, false, false);
      }
      // D layout: lane holds col n = lane&15; VGPR vr holds row vr + 8*hi.
#pragma unroll
      for (int vr = 0; vr < 8; ++vr) {
        int row = (rt << 4) + vr + (hi << 3);
        m = fmaxf(m, fmaxf(c[vr], 0.0f) * sValid[row]);
      }
    }
    m = fmaxf(m, __shfl_xor(m, 16, 32));   // fold the two 8-row half-tiles
    if (lane < 16) pooled[(size_t)kp * 256 + ct * 16 + ln] = m;
  }
}

// ---------------------------------------------------------------------------
// 5) Fuse: [B*K, 512] (= concat pooled_a|pooled_b) @ [512,128] + relu,
//    prepend keypoint xyz. 16 rows per block, 8 waves = 8 col tiles,
//    16 WMMA k-steps per wave.
// ---------------------------------------------------------------------------
__global__ __launch_bounds__(256)
void fuse_kernel(const float* __restrict__ pooled_a, const float* __restrict__ pooled_b,
                 const _Float16* __restrict__ wf_frag, const float* __restrict__ kp_xyz,
                 float* __restrict__ out) {
  const int rowbase = blockIdx.x * 16;
  const int t = threadIdx.x, lane = t & 31, wid = t >> 5;
  const int hi = lane >> 4, ln = lane & 15;
  const int mrow = rowbase + ln;           // A row this lane supplies
  const int ct = wid;                      // 8 col tiles -> 128 cols

  v8f c = {};
#pragma unroll
  for (int ks = 0; ks < 16; ++ks) {
    int k0 = ks * 32 + hi * 16;            // halves 0..7
    int k1 = k0 + 8;                       // halves 8..15
    const float* s0 = (k0 < 256) ? &pooled_a[(size_t)mrow * 256 + k0]
                                 : &pooled_b[(size_t)mrow * 256 + (k0 - 256)];
    const float* s1 = (k1 < 256) ? &pooled_a[(size_t)mrow * 256 + k1]
                                 : &pooled_b[(size_t)mrow * 256 + (k1 - 256)];
    v16h a;
#pragma unroll
    for (int e = 0; e < 8; ++e) { a[e] = (_Float16)s0[e]; a[8 + e] = (_Float16)s1[e]; }
    v16h bf = *(const v16h*)(wf_frag + ((size_t)(ks * 8 + ct) * 32 + lane) * 16);
    c = __builtin_amdgcn_wmma_f32_16x16x32_f16(
        false, a, false, bf, (short)0, c, false, false);
  }
#pragma unroll
  for (int vr = 0; vr < 8; ++vr) {
    int row = rowbase + vr + (hi << 3);
    out[(size_t)row * OUT_STRIDE + 3 + ct * 16 + ln] = fmaxf(c[vr], 0.0f);
  }
  if (t < 48) {
    int row = rowbase + t / 3, cidx = t % 3;
    out[(size_t)row * OUT_STRIDE + cidx] = kp_xyz[(size_t)row * 3 + cidx];
  }
}

// ---------------------------------------------------------------------------
extern "C" void kernel_launch(void* const* d_in, const int* in_sizes, int n_in,
                              void* d_out, int out_size, void* d_ws, size_t ws_size,
                              hipStream_t stream) {
  const float* clouds = (const float*)d_in[0];
  const float* w1_a   = (const float*)d_in[1];
  const float* w2_a   = (const float*)d_in[2];
  const float* w1_b   = (const float*)d_in[3];
  const float* w2_b   = (const float*)d_in[4];
  const float* w_fuse = (const float*)d_in[5];

  const int N = N_PTS;
  const int K = N_KP;
  const int B = in_sizes[0] / (4 * N);
  const int BK = B * K;

  char* ws = (char*)d_ws;
  size_t off = 0;
  auto alloc = [&](size_t bytes) -> void* {
    void* p = ws + off;
    off = (off + bytes + 255) & ~(size_t)255;
    return p;
  };
  int*      kp_idx    = (int*)      alloc((size_t)BK * 4);
  float*    kp_xyz    = (float*)    alloc((size_t)BK * 3 * 4);
  int*      neigh_idx = (int*)      alloc((size_t)BK * 32 * 4);
  float*    neigh_d2  = (float*)    alloc((size_t)BK * 32 * 4);
  _Float16* w2a_frag  = (_Float16*) alloc((size_t)128 * 256 * 2);
  _Float16* w2b_frag  = (_Float16*) alloc((size_t)128 * 256 * 2);
  _Float16* wf_frag   = (_Float16*) alloc((size_t)512 * 128 * 2);
  float*    pooled_a  = (float*)    alloc((size_t)BK * 256 * 4);
  float*    pooled_b  = (float*)    alloc((size_t)BK * 256 * 4);
  (void)ws_size; (void)n_in; (void)out_size;

  fps_kernel<<<dim3(B), dim3(FPS_THREADS), 0, stream>>>(clouds, N, K, kp_idx, kp_xyz);

  knn_kernel<<<dim3((BK + 255) / 256), dim3(256), 0, stream>>>(
      clouds, kp_xyz, N, K, neigh_idx, neigh_d2);

  pack_w_kernel<<<dim3((128 * 256 + 255) / 256), dim3(256), 0, stream>>>(w2_a, w2a_frag, 128, 256);
  pack_w_kernel<<<dim3((128 * 256 + 255) / 256), dim3(256), 0, stream>>>(w2_b, w2b_frag, 128, 256);
  pack_w_kernel<<<dim3((512 * 128 + 255) / 256), dim3(256), 0, stream>>>(w_fuse, wf_frag, 512, 128);

  sa_kernel<<<dim3(BK), dim3(256), 0, stream>>>(
      clouds, kp_xyz, neigh_idx, neigh_d2, w1_a, w2a_frag, pooled_a, N, K, 16, 0.8f * 0.8f);
  sa_kernel<<<dim3(BK), dim3(256), 0, stream>>>(
      clouds, kp_xyz, neigh_idx, neigh_d2, w1_b, w2b_frag, pooled_b, N, K, 32, 1.6f * 1.6f);

  fuse_kernel<<<dim3(BK / 16), dim3(256), 0, stream>>>(
      pooled_a, pooled_b, wf_frag, kp_xyz, (float*)d_out);
}
